// SpatialAttentionModule_41704132444430
// MI455X (gfx1250) — matrix operand order repaired and checked by reading, not compiled, not run on hardware
//
#include <hip/hip_runtime.h>
#include <hip/hip_bf16.h>

#define B_ 4
#define N_ 2048
#define H_ 128
#define HEADS_ 4
#define HD_ 32
#define R_ 8
#define DK_ 64     // padded K dim: 32 (q.k) + 8 (u.v) + 24 zero
#define CT_ 64     // column tile of keys per iteration
#define LOG2E_ 1.4426950408889634f

typedef __attribute__((ext_vector_type(16))) __bf16 v16bf;
typedef __attribute__((ext_vector_type(8)))  float  v8f;

union Frag { v16bf v; unsigned u[8]; };

static __device__ __forceinline__ float hmax16(float v) {
  v = fmaxf(v, __shfl_xor(v, 1, 32));
  v = fmaxf(v, __shfl_xor(v, 2, 32));
  v = fmaxf(v, __shfl_xor(v, 4, 32));
  v = fmaxf(v, __shfl_xor(v, 8, 32));
  return v;
}
static __device__ __forceinline__ float hsum16(float v) {
  v += __shfl_xor(v, 1, 32);
  v += __shfl_xor(v, 2, 32);
  v += __shfl_xor(v, 4, 32);
  v += __shfl_xor(v, 8, 32);
  return v;
}

// ---------------------------------------------------------------------------
// Kernel 1: low-rank QKV projection, build padded bf16 Qe/Ke and transposed V.
// One block per (b,n) row, 128 threads.
// ---------------------------------------------------------------------------
__global__ void prep_qkv(const float* __restrict__ x,
                         const float* __restrict__ Wql, const float* __restrict__ bql,
                         const float* __restrict__ Wqh, const float* __restrict__ bqh,
                         const float* __restrict__ uu,  const float* __restrict__ vv,
                         __bf16* __restrict__ Qe, __bf16* __restrict__ Ke,
                         __bf16* __restrict__ Vt)
{
  __shared__ float xs[H_];
  __shared__ float ts[3 * R_];
  const int row = blockIdx.x;              // b*N + n
  const int b = row >> 11, n = row & (N_ - 1);
  const int c = threadIdx.x;               // 0..127
  xs[c] = x[(size_t)row * H_ + c];
  __syncthreads();
  if (c < 3 * R_) {
    float acc = bql[c];
#pragma unroll 8
    for (int i = 0; i < H_; ++i) acc += xs[i] * Wql[i * (3 * R_) + c];
    ts[c] = acc;
  }
  __syncthreads();
  const int h = c >> 5, d = c & 31;
  float q = bqh[c], k = bqh[H_ + c], w = bqh[2 * H_ + c];
#pragma unroll
  for (int j = 0; j < 3 * R_; ++j) {
    const float t = ts[j];
    q += t * Wqh[j * (3 * H_) + c];
    k += t * Wqh[j * (3 * H_) + H_ + c];
    w += t * Wqh[j * (3 * H_) + 2 * H_ + c];
  }
  const int bh = b * HEADS_ + h;
  Qe[((size_t)bh * N_ + n) * DK_ + d] = (__bf16)(q * 0.17677669529663687f); // 1/sqrt(32)
  Ke[((size_t)bh * N_ + n) * DK_ + d] = (__bf16)k;
  Vt[((size_t)bh * HD_ + d) * N_ + n] = (__bf16)w;   // transposed: [bh][d][n]
  // graph-bias extension (K = 32..39) and zero padding (K = 40..63)
  if (c < HEADS_ * R_) {
    const int hh = c >> 3, r = c & 7;
    const int bhh = b * HEADS_ + hh;
    Qe[((size_t)bhh * N_ + n) * DK_ + 32 + r] = (__bf16)uu[((size_t)hh * N_ + n) * R_ + r];
    Ke[((size_t)bhh * N_ + n) * DK_ + 32 + r] = (__bf16)vv[((size_t)hh * R_ + r) * N_ + n];
  }
  if (c < HEADS_ * 24) {
    const int hh = c / 24, kk = 40 + c % 24;
    const int bhh = b * HEADS_ + hh;
    Qe[((size_t)bhh * N_ + n) * DK_ + kk] = (__bf16)0.0f;
    Ke[((size_t)bhh * N_ + n) * DK_ + kk] = (__bf16)0.0f;
  }
}

// ---------------------------------------------------------------------------
// Kernel 2: adjT[m][n] = softplus(adj_scale) * adj_prior[n][m]  (fp32, tiled
// LDS transpose, coalesced load and store). 32x32 tiles, 256 threads.
// ---------------------------------------------------------------------------
__global__ void prep_adjT(const float* __restrict__ adj,
                          const float* __restrict__ adj_scale,
                          float* __restrict__ adjT)
{
  __shared__ float tile[32][33];
  const float scale = log1pf(expf(adj_scale[0]));   // softplus
  const int bx = blockIdx.x & (N_ / 32 - 1);        // m tile
  const int by = blockIdx.x / (N_ / 32);            // n tile
  const int tx = threadIdx.x & 31;
  const int ty = threadIdx.x >> 5;                  // 0..7
#pragma unroll
  for (int k = 0; k < 4; ++k) {
    const int n = by * 32 + ty + 8 * k;
    tile[ty + 8 * k][tx] = scale * adj[(size_t)n * N_ + bx * 32 + tx];
  }
  __syncthreads();
#pragma unroll
  for (int k = 0; k < 4; ++k) {
    const int m = bx * 32 + ty + 8 * k;
    adjT[(size_t)m * N_ + by * 32 + tx] = tile[tx][ty + 8 * k];
  }
}

// ---------------------------------------------------------------------------
// Kernel 3: flash attention over N=2048 with WMMA bf16 (f32 accum).
// Grid: B*HEADS*(N/128) blocks, 8 waves/block, 16 query rows per wave.
// ---------------------------------------------------------------------------
__global__ void __launch_bounds__(256) attn_kernel(
    const __bf16* __restrict__ Qe, const __bf16* __restrict__ Ke,
    const __bf16* __restrict__ Vt, const float* __restrict__ adjT,
    float* __restrict__ attn_out)
{
  __shared__ __bf16 Pl[8][16 * CT_];       // per-wave P staging (C-layout -> A-layout)
  const int bh   = blockIdx.x >> 4;        // 16 row-blocks per (b,h)
  const int rb   = blockIdx.x & 15;
  const int b    = bh >> 2, h = bh & 3;
  const int wave = threadIdx.x >> 5;
  const int lane = threadIdx.x & 31;
  const int hf   = lane >> 4, l16 = lane & 15;
  const int qrow0 = rb * 128 + wave * 16;

  // Q A-fragments for K=0..31 and K=32..63 (contiguous: K = elem + 16*half)
  Frag aq0, aq1;
  {
    const __bf16* qr = Qe + ((size_t)bh * N_ + (qrow0 + l16)) * DK_;
#pragma unroll
    for (int j = 0; j < 8; ++j) {
      const int kb = 2 * j + 16 * hf;
      aq0.u[j] = *(const unsigned*)(qr + kb);
      aq1.u[j] = *(const unsigned*)(qr + kb + 32);
    }
  }

  v8f o0 = {}, o1 = {};
  float m_run[8], l_run[8];
#pragma unroll
  for (int i = 0; i < 8; ++i) { m_run[i] = -3.0e38f; l_run[i] = 0.0f; }

  const __bf16* Kbase = Ke + (size_t)bh * N_ * DK_;
  const __bf16* Vbase = Vt + (size_t)bh * HD_ * N_;
  // per-lane bias column base: 8 consecutive rows (i + 8*hf) at fixed col m
  const float* Abase = adjT + qrow0 + 8 * hf;

  for (int jt = 0; jt < N_ / CT_; ++jt) {
    const int jbase = jt * CT_;
    v8f s[4];
#pragma unroll
    for (int st = 0; st < 4; ++st) {
      // K B-fragments: contiguous 32B per lane (B layout: K = elem + 16*half)
      const __bf16* kr = Kbase + (size_t)(jbase + st * 16 + l16) * DK_ + 16 * hf;
      const v16bf kb0 = *(const v16bf*)kr;
      const v16bf kb1 = *(const v16bf*)(kr + 32);
      v8f acc = {};
      acc = __builtin_amdgcn_wmma_f32_16x16x32_bf16(false, aq0.v, false, kb0,
                                                    (short)0, acc, false, false);
      acc = __builtin_amdgcn_wmma_f32_16x16x32_bf16(false, aq1.v, false, kb1,
                                                    (short)0, acc, false, false);
      // adjacency bias: 8 consecutive fp32 per lane (transposed layout)
      const int mcol = jbase + st * 16 + l16;
      const v8f bias = *(const v8f*)(Abase + (size_t)mcol * N_);
      s[st] = acc + bias;
    }
    // online softmax update (rows live across 16-lane halves)
    float alpha[8], psum[8];
#pragma unroll
    for (int i = 0; i < 8; ++i) {
      float v = fmaxf(fmaxf(s[0][i], s[1][i]), fmaxf(s[2][i], s[3][i]));
      v = hmax16(v);
      const float mn = fmaxf(m_run[i], v);
      alpha[i] = exp2f((m_run[i] - mn) * LOG2E_);
      m_run[i] = mn;
      psum[i] = 0.0f;
    }
    __bf16* pw = &Pl[wave][0];
#pragma unroll
    for (int st = 0; st < 4; ++st) {
#pragma unroll
      for (int i = 0; i < 8; ++i) {
        const float p = exp2f((s[st][i] - m_run[i]) * LOG2E_);
        psum[i] += p;
        pw[(i + 8 * hf) * CT_ + st * 16 + l16] = (__bf16)p;
      }
    }
#pragma unroll
    for (int i = 0; i < 8; ++i) {
      l_run[i] = l_run[i] * alpha[i] + hsum16(psum[i]);
      o0[i] *= alpha[i];
      o1[i] *= alpha[i];
    }
    // reload P tile as A-fragments (same-wave DS ops are in-order: no barrier)
    Frag pa0, pa1;
    {
      const __bf16* pr = pw + l16 * CT_;
#pragma unroll
      for (int j = 0; j < 8; ++j) {
        const int kb = 2 * j + 16 * hf;
        pa0.u[j] = *(const unsigned*)(pr + kb);
        pa1.u[j] = *(const unsigned*)(pr + kb + 32);
      }
    }
    // V B-fragments from transposed V (contiguous 32B per lane) + accumulate
    {
      const __bf16* vr0 = Vbase + (size_t)l16 * N_ + jbase + 16 * hf;
      const v16bf vb00 = *(const v16bf*)vr0;
      const v16bf vb01 = *(const v16bf*)(vr0 + 32);
      o0 = __builtin_amdgcn_wmma_f32_16x16x32_bf16(false, pa0.v, false, vb00,
                                                   (short)0, o0, false, false);
      o0 = __builtin_amdgcn_wmma_f32_16x16x32_bf16(false, pa1.v, false, vb01,
                                                   (short)0, o0, false, false);
      const __bf16* vr1 = Vbase + (size_t)(16 + l16) * N_ + jbase + 16 * hf;
      const v16bf vb10 = *(const v16bf*)vr1;
      const v16bf vb11 = *(const v16bf*)(vr1 + 32);
      o1 = __builtin_amdgcn_wmma_f32_16x16x32_bf16(false, pa0.v, false, vb10,
                                                   (short)0, o1, false, false);
      o1 = __builtin_amdgcn_wmma_f32_16x16x32_bf16(false, pa1.v, false, vb11,
                                                   (short)0, o1, false, false);
    }
    if (jt + 1 < N_ / CT_) {
      __builtin_prefetch(Kbase + (size_t)(jbase + CT_ + l16) * DK_, 0, 1);  // global_prefetch_b8
      __builtin_prefetch(Abase + (size_t)(jbase + CT_ + l16) * N_, 0, 1);
    }
  }
  // finalize: divide by row sum, scatter to [b][n][h*32+d]
#pragma unroll
  for (int i = 0; i < 8; ++i) {
    const float inv = 1.0f / l_run[i];
    const int n = qrow0 + i + 8 * hf;
    float* orow = attn_out + ((size_t)b * N_ + n) * H_ + h * HD_;
    orow[l16]      = o0[i] * inv;
    orow[16 + l16] = o1[i] * inv;
  }
}

// ---------------------------------------------------------------------------
// Kernel 4: low-rank output projection + residual + LayerNorm (+ reg_loss).
// One block per (b,n) row, 128 threads.
// ---------------------------------------------------------------------------
__global__ void final_kernel(const float* __restrict__ ao,
                             const float* __restrict__ x,
                             const float* __restrict__ Wol, const float* __restrict__ bol,
                             const float* __restrict__ Woh, const float* __restrict__ boh,
                             const float* __restrict__ gamma, const float* __restrict__ beta,
                             const float* __restrict__ log_reg_w,
                             float* __restrict__ out)
{
  __shared__ float aos[H_];
  __shared__ float ts[R_];
  __shared__ float red[H_];
  const int row = blockIdx.x;
  const int c = threadIdx.x;
  aos[c] = ao[(size_t)row * H_ + c];
  __syncthreads();
  if (c < R_) {
    float acc = bol[c];
#pragma unroll 8
    for (int i = 0; i < H_; ++i) acc += aos[i] * Wol[i * R_ + c];
    ts[c] = acc;
  }
  __syncthreads();
  float o = boh[c];
#pragma unroll
  for (int r = 0; r < R_; ++r) o += ts[r] * Woh[r * H_ + c];
  const float y0 = o + x[(size_t)row * H_ + c];
  red[c] = y0;
  __syncthreads();
  for (int s2 = H_ / 2; s2 > 0; s2 >>= 1) { if (c < s2) red[c] += red[c + s2]; __syncthreads(); }
  const float mean = red[0] * (1.0f / H_);
  __syncthreads();
  const float dcen = y0 - mean;
  red[c] = dcen * dcen;
  __syncthreads();
  for (int s2 = H_ / 2; s2 > 0; s2 >>= 1) { if (c < s2) red[c] += red[c + s2]; __syncthreads(); }
  const float rstd = rsqrtf(red[0] * (1.0f / H_) + 1e-5f);
  out[(size_t)row * H_ + c] = dcen * rstd * gamma[c] + beta[c];
  // reg_loss = exp(log_reg_w) * mean(|softmax|) = exp(log_reg_w)/N exactly
  if (row == 0 && c == 0) out[(size_t)B_ * N_ * H_] = expf(log_reg_w[0]) * (1.0f / N_);
}

// ---------------------------------------------------------------------------
extern "C" void kernel_launch(void* const* d_in, const int* in_sizes, int n_in,
                              void* d_out, int out_size, void* d_ws, size_t ws_size,
                              hipStream_t stream)
{
  (void)in_sizes; (void)n_in; (void)out_size; (void)ws_size;
  const float* x         = (const float*)d_in[0];
  const float* Wql       = (const float*)d_in[1];
  const float* bql       = (const float*)d_in[2];
  const float* Wqh       = (const float*)d_in[3];
  const float* bqh       = (const float*)d_in[4];
  const float* uu        = (const float*)d_in[5];
  const float* vv        = (const float*)d_in[6];
  const float* adj_scale = (const float*)d_in[7];
  const float* Wol       = (const float*)d_in[8];
  const float* bol       = (const float*)d_in[9];
  const float* Woh       = (const float*)d_in[10];
  const float* boh       = (const float*)d_in[11];
  const float* gamma     = (const float*)d_in[12];
  const float* beta      = (const float*)d_in[13];
  const float* lrw       = (const float*)d_in[14];
  const float* adj       = (const float*)d_in[15];

  char* ws = (char*)d_ws;
  __bf16* Qe   = (__bf16*)(ws);                // 4 MB
  __bf16* Ke   = (__bf16*)(ws + 4194304);      // 4 MB
  __bf16* Vt   = (__bf16*)(ws + 8388608);      // 2 MB
  float*  adjT = (float*) (ws + 10485760);     // 16 MB (fp32, transposed, scaled)
  float*  ao   = (float*) (ws + 27262976);     // 4 MB

  prep_qkv<<<dim3(B_ * N_), dim3(H_), 0, stream>>>(x, Wql, bql, Wqh, bqh, uu, vv,
                                                   Qe, Ke, Vt);
  prep_adjT<<<dim3((N_ / 32) * (N_ / 32)), dim3(256), 0, stream>>>(adj, adj_scale,
                                                                   adjT);
  attn_kernel<<<dim3(B_ * HEADS_ * (N_ / 128)), dim3(256), 0, stream>>>(Qe, Ke, Vt,
                                                                        adjT, ao);
  final_kernel<<<dim3(B_ * N_), dim3(H_), 0, stream>>>(ao, x, Wol, bol, Woh, boh,
                                                       gamma, beta, lrw,
                                                       (float*)d_out);
}
